// GatedQueryAttLayer_44427141710076
// MI455X (gfx1250) — matrix operand order repaired and checked by reading, not compiled, not run on hardware
//
#include <hip/hip_runtime.h>

// ---------------------------------------------------------------------------
// GatedQueryAttLayer for MI455X (gfx1250, wave32, WMMA bf16 path)
// B=16 S=512 E=1024 H=16 DK=64
// ---------------------------------------------------------------------------

typedef __attribute__((ext_vector_type(2)))  __bf16 v2bf;
typedef __attribute__((ext_vector_type(8)))  float  v8f;
typedef __attribute__((ext_vector_type(8)))  __bf16 v8bf;
typedef __attribute__((ext_vector_type(16))) __bf16 v16bf;

#define NB  16
#define NS  512
#define NE  1024
#define NH  16
#define NDK 64

// ---- bf16 <-> f32 helpers (hw cvt on gfx1250) -----------------------------
static __device__ __forceinline__ __bf16 f2bf(float f) {
  return static_cast<__bf16>(f);
}
// pack two f32 -> 2xbf16 in one dword (lowers to v_cvt_pk_bf16_f32)
static __device__ __forceinline__ unsigned pk2bf(float a, float b) {
  v2bf r;
  r[0] = static_cast<__bf16>(a);
  r[1] = static_cast<__bf16>(b);
  return __builtin_bit_cast(unsigned, r);
}
static __device__ __forceinline__ float lo16f(unsigned u) {  // bf16 in [15:0]
  return __builtin_bit_cast(float, u << 16);
}
static __device__ __forceinline__ float hi16f(unsigned u) {  // bf16 in [31:16]
  return __builtin_bit_cast(float, u & 0xFFFF0000u);
}
static __device__ __forceinline__ v8f vzero8() {
  v8f z;
#pragma unroll
  for (int i = 0; i < 8; ++i) z[i] = 0.f;
  return z;
}

// ---- WMMA fragment loaders (contiguous k = 16*(lane/16)+s in A and B, wave32) -------------------
// A fragment: 16x32 tile, row-major source with leading dim `ld` (elements).
// lanes 0-15 : row = lane,    K = 0..15
// lanes 16-31: row = lane-16, K = 16..31
static __device__ __forceinline__ v16bf load_a_frag(const __bf16* base, int ld, int lane) {
  int row  = lane & 15;
  int coff = (lane < 16) ? 0 : 16;
  const __bf16* p = base + row * ld + coff;
  v8bf lo = *(const v8bf*)(p);
  v8bf hi = *(const v8bf*)(p + 8);
  v16bf r;
#pragma unroll
  for (int i = 0; i < 8; ++i) { r[i] = lo[i]; r[i + 8] = hi[i]; }
  return r;
}
// B fragment: 32x16 tile from a COLUMN-MAJOR buffer colmaj[n][k], ld = k-stride.
// lanes 0-15 : col = lane,    K = 0..15 (one contiguous 32B load)
// lanes 16-31: col = lane-16, K = 16..31
static __device__ __forceinline__ v16bf load_b_frag(const __bf16* base, int ld, int lane) {
  int col  = lane & 15;
  int koff = (lane < 16) ? 0 : 16;
  return *(const v16bf*)(base + col * ld + koff);
}
static __device__ __forceinline__ v8f wmma_bf16(v16bf a, v16bf b, v8f c) {
  return __builtin_amdgcn_wmma_f32_16x16x32_bf16(false, a, false, b, (short)0, c,
                                                 false, false);
}
// reductions over the 16-lane half that holds one C-tile row (wave32)
static __device__ __forceinline__ float hmax16(float v) {
#pragma unroll
  for (int m = 8; m >= 1; m >>= 1) v = fmaxf(v, __shfl_xor(v, m, 32));
  return v;
}
static __device__ __forceinline__ float hsum16(float v) {
#pragma unroll
  for (int m = 8; m >= 1; m >>= 1) v += __shfl_xor(v, m, 32);
  return v;
}

// ---------------------------------------------------------------------------
// Kernel 1: QKV projection GEMM.  out = inp @ W + b  (M=8192, N=1024, K=1024)
// grid.z selects {Wq,Wk,Wv}. q,k stored bf16 [B,H,S,DK]; v stored TRANSPOSED
// bf16 [B,H,DK,S] so attention's P@V B-fragments are contiguous loads.
// Workgroup tile 128x128, 8 waves of 64x32, K-step 32. DOUBLE-BUFFERED LDS:
// tile i+1 is staged (packed uint4 -> ds_store_b128, v_cvt_pk_bf16_f32)
// between the single per-iteration barrier and the WMMAs of tile i.
// ---------------------------------------------------------------------------
__global__ __launch_bounds__(256) void qkv_proj_kernel(
    const float* __restrict__ inp,
    const float* __restrict__ Wq, const float* __restrict__ bq,
    const float* __restrict__ Wk, const float* __restrict__ bk,
    const float* __restrict__ Wv, const float* __restrict__ bv,
    __bf16* __restrict__ qbuf, __bf16* __restrict__ kbuf,
    __bf16* __restrict__ vT) {
  __shared__ __align__(32) unsigned As_u[2][128][16];  // 128 x 32 bf16, row-major
  __shared__ __align__(32) unsigned Bs_u[2][128][16];  // column-major W: [n][k]
  const int tid  = threadIdx.x;
  const int wave = tid >> 5, lane = tid & 31;
  const int wm = wave >> 2, wn = wave & 3;             // 2x4 wave grid
  const int m0 = blockIdx.x * 128, n0 = blockIdx.y * 128;
  const int z  = blockIdx.z;
  const float* W    = (z == 0) ? Wq : (z == 1) ? Wk : Wv;
  const float* bias = (z == 0) ? bq : (z == 1) ? bk : bv;

  // staging assignments (constant per thread)
  const int ar = tid >> 1, ac8 = (tid & 1) * 8;        // A: row, uint-col base
  const int bn = tid & 127, bkh = (tid >> 7) * 16;     // B: column, k base

  auto stage = [&](int buf, int k0) {
    {  // A: 16 contiguous f32 -> 8 packed dwords -> 2x b128 LDS stores
      const float* src = inp + (size_t)(m0 + ar) * NE + k0 + ac8 * 2;
      unsigned p[8];
#pragma unroll
      for (int j = 0; j < 8; ++j) p[j] = pk2bf(src[2 * j], src[2 * j + 1]);
      uint4* dst = (uint4*)&As_u[buf][ar][ac8];
      dst[0] = make_uint4(p[0], p[1], p[2], p[3]);
      dst[1] = make_uint4(p[4], p[5], p[6], p[7]);
    }
    {  // B transposed: one column per thread (reads coalesced over n)
      const float* wp = W + (size_t)(k0 + bkh) * NE + n0 + bn;
      unsigned p[8];
#pragma unroll
      for (int j = 0; j < 8; ++j)
        p[j] = pk2bf(wp[(size_t)(2 * j) * NE], wp[(size_t)(2 * j + 1) * NE]);
      uint4* dst = (uint4*)&Bs_u[buf][bn][bkh >> 1];
      dst[0] = make_uint4(p[0], p[1], p[2], p[3]);
      dst[1] = make_uint4(p[4], p[5], p[6], p[7]);
    }
  };

  v8f acc[4][2];
#pragma unroll
  for (int i = 0; i < 4; ++i)
#pragma unroll
    for (int j = 0; j < 2; ++j) acc[i][j] = vzero8();

  stage(0, 0);
  for (int k0 = 0; k0 < NE; k0 += 32) {
    const int buf = (k0 >> 5) & 1;
    __syncthreads();                      // staged tile `buf` now visible
    if (k0 + 32 < NE) stage(buf ^ 1, k0 + 32);  // overlap next-tile staging

    const __bf16* Asb = (const __bf16*)As_u[buf];
    const __bf16* Bsb = (const __bf16*)Bs_u[buf];
    v16bf af[4], bfr[2];
#pragma unroll
    for (int mt = 0; mt < 4; ++mt)
      af[mt] = load_a_frag(Asb + (wm * 64 + mt * 16) * 32, 32, lane);
#pragma unroll
    for (int nt = 0; nt < 2; ++nt)
      bfr[nt] = load_b_frag(Bsb + (wn * 32 + nt * 16) * 32, 32, lane);
#pragma unroll
    for (int mt = 0; mt < 4; ++mt)
#pragma unroll
      for (int nt = 0; nt < 2; ++nt)
        acc[mt][nt] = wmma_bf16(af[mt], bfr[nt], acc[mt][nt]);
  }

  // epilogue: bias add + scatter into head-split layouts
  const int lh  = lane & 15;
  const int rof = (lane < 16) ? 0 : 8;
#pragma unroll
  for (int mt = 0; mt < 4; ++mt) {
#pragma unroll
    for (int nt = 0; nt < 2; ++nt) {
      int gn = n0 + wn * 32 + nt * 16 + lh;
      float bia = bias[gn];
      int h = gn >> 6, d = gn & 63;
#pragma unroll
      for (int vg = 0; vg < 8; ++vg) {
        int gm = m0 + wm * 64 + mt * 16 + vg + rof;
        int bb = gm >> 9, s = gm & 511;
        float val = acc[mt][nt][vg] + bia;
        size_t bh = (size_t)bb * NH + h;
        if (z == 0)      qbuf[(bh * NS + s) * NDK + d] = f2bf(val);
        else if (z == 1) kbuf[(bh * NS + s) * NDK + d] = f2bf(val);
        else             vT[(bh * NDK + d) * NS + s]   = f2bf(val);
      }
    }
  }
}

// ---------------------------------------------------------------------------
// Kernel 2: per-head gating, all in WMMA.
//   G = (q@Wfq+bfq)*(k@Wfk+bfk);  M = sigmoid(G@Wfg+bfg)
//   q *= M[:, :DK];  k *= M[:, DK:]   (in place)
// One wave per 16-row s-tile; weights staged column-major (packed) in LDS.
// ---------------------------------------------------------------------------
__global__ __launch_bounds__(256) void gate_kernel(
    const float* __restrict__ Wfq, const float* __restrict__ bfq,
    const float* __restrict__ Wfk, const float* __restrict__ bfk,
    const float* __restrict__ Wfg, const float* __restrict__ bfg,
    __bf16* __restrict__ qbuf, __bf16* __restrict__ kbuf) {
  __shared__ __align__(32) unsigned WfqTu[64][32];   // [n][k] 64x64 bf16
  __shared__ __align__(32) unsigned WfkTu[64][32];
  __shared__ __align__(32) unsigned WfgTu[128][32];  // [n][k] 128x64 bf16
  __shared__ __align__(32) __bf16 g_lds[8][16][64];
  __shared__ __align__(32) __bf16 m_lds[8][16][128];
  const __bf16* WfqT = (const __bf16*)WfqTu;
  const __bf16* WfkT = (const __bf16*)WfkTu;
  const __bf16* WfgT = (const __bf16*)WfgTu;
  const int tid  = threadIdx.x;
  const int wave = tid >> 5, lane = tid & 31;
  const int lh = lane & 15, rof = (lane < 16) ? 0 : 8;

  {  // stage 64x64 weights transposed, packed; reads coalesced over n
    int n = tid & 63, kh = (tid >> 6) * 16;
    unsigned pq[8], pk[8];
#pragma unroll
    for (int j = 0; j < 8; ++j) {
      pq[j] = pk2bf(Wfq[(kh + 2 * j) * 64 + n], Wfq[(kh + 2 * j + 1) * 64 + n]);
      pk[j] = pk2bf(Wfk[(kh + 2 * j) * 64 + n], Wfk[(kh + 2 * j + 1) * 64 + n]);
    }
    uint4* dq = (uint4*)&WfqTu[n][kh >> 1];
    dq[0] = make_uint4(pq[0], pq[1], pq[2], pq[3]);
    dq[1] = make_uint4(pq[4], pq[5], pq[6], pq[7]);
    uint4* dk = (uint4*)&WfkTu[n][kh >> 1];
    dk[0] = make_uint4(pk[0], pk[1], pk[2], pk[3]);
    dk[1] = make_uint4(pk[4], pk[5], pk[6], pk[7]);
  }
  {  // stage 64x128 Wfg transposed, packed
    int n = tid & 127, kh = (tid >> 7) * 32;
    unsigned p[16];
#pragma unroll
    for (int j = 0; j < 16; ++j)
      p[j] = pk2bf(Wfg[(kh + 2 * j) * 128 + n], Wfg[(kh + 2 * j + 1) * 128 + n]);
    uint4* dg = (uint4*)&WfgTu[n][kh >> 1];
#pragma unroll
    for (int q = 0; q < 4; ++q)
      dg[q] = make_uint4(p[4 * q], p[4 * q + 1], p[4 * q + 2], p[4 * q + 3]);
  }
  __syncthreads();

  const int tileIdx = blockIdx.x * 8 + wave;   // 8192 tiles total
  const int bh = tileIdx >> 5;                 // 32 s-tiles per (b,h)
  const int st = (tileIdx & 31) << 4;
  __bf16* qrow = qbuf + ((size_t)bh * NS + st) * NDK;
  __bf16* krow = kbuf + ((size_t)bh * NS + st) * NDK;

  v16bf aQ[2], aK[2];
#pragma unroll
  for (int kc = 0; kc < 2; ++kc) {
    aQ[kc] = load_a_frag(qrow + kc * 32, NDK, lane);
    aK[kc] = load_a_frag(krow + kc * 32, NDK, lane);
  }
#pragma unroll
  for (int nt = 0; nt < 4; ++nt) {
    v8f gq = vzero8(), gk = vzero8();
#pragma unroll
    for (int kc = 0; kc < 2; ++kc) {
      gq = wmma_bf16(aQ[kc], load_b_frag(WfqT + (nt * 16) * 64 + kc * 32, 64, lane), gq);
      gk = wmma_bf16(aK[kc], load_b_frag(WfkT + (nt * 16) * 64 + kc * 32, 64, lane), gk);
    }
    float bq_ = bfq[nt * 16 + lh], bk_ = bfk[nt * 16 + lh];
#pragma unroll
    for (int vg = 0; vg < 8; ++vg)
      g_lds[wave][vg + rof][nt * 16 + lh] = f2bf((gq[vg] + bq_) * (gk[vg] + bk_));
  }
  __syncthreads();

  v16bf aG[2];
#pragma unroll
  for (int kc = 0; kc < 2; ++kc)
    aG[kc] = load_a_frag(&g_lds[wave][0][kc * 32], 64, lane);
#pragma unroll
  for (int nt = 0; nt < 8; ++nt) {
    v8f mm = vzero8();
#pragma unroll
    for (int kc = 0; kc < 2; ++kc)
      mm = wmma_bf16(aG[kc], load_b_frag(WfgT + (nt * 16) * 64 + kc * 32, 64, lane), mm);
    float bg = bfg[nt * 16 + lh];
#pragma unroll
    for (int vg = 0; vg < 8; ++vg) {
      float sig = 1.f / (1.f + __expf(-(mm[vg] + bg)));
      m_lds[wave][vg + rof][nt * 16 + lh] = f2bf(sig);
    }
  }
  __syncthreads();

  // in-place modulation, 2 elements per dword (this wave owns these rows)
  unsigned* qu = (unsigned*)qrow;
  unsigned* ku = (unsigned*)krow;
  const unsigned* mu = (const unsigned*)&m_lds[wave][0][0];  // row stride 64 dwords
  for (int e = lane; e < 16 * NDK / 2; e += 32) {
    int r = e >> 5, dp = e & 31;
    unsigned q2 = qu[r * 32 + dp];
    unsigned k2 = ku[r * 32 + dp];
    unsigned mq = mu[r * 64 + dp];
    unsigned mk = mu[r * 64 + 32 + dp];
    qu[r * 32 + dp] = pk2bf(lo16f(q2) * lo16f(mq), hi16f(q2) * hi16f(mq));
    ku[r * 32 + dp] = pk2bf(lo16f(k2) * lo16f(mk), hi16f(k2) * hi16f(mk));
  }
}

// ---------------------------------------------------------------------------
// Kernel 3: flash attention per (b,h). One wave owns 16 query rows; K-blocks
// of 64. Scores never touch HBM. P goes C-layout -> LDS -> A-layout for P@V.
// V is pre-transposed so B-fragments are single contiguous global loads.
// ---------------------------------------------------------------------------
__global__ __launch_bounds__(256) void attn_kernel(
    const __bf16* __restrict__ qm, const __bf16* __restrict__ km,
    const __bf16* __restrict__ vT, float* __restrict__ out) {
  __shared__ __align__(32) __bf16 p_lds[8][16][64];
  const int tid  = threadIdx.x;
  const int wave = tid >> 5, lane = tid & 31;
  const int bh = blockIdx.x >> 2;                     // (b,h)
  const int q0 = ((blockIdx.x & 3) << 7) + wave * 16; // this wave's 16 q rows
  const int lh = lane & 15, rof = (lane < 16) ? 0 : 8;

  v16bf aQ[2];
#pragma unroll
  for (int kc = 0; kc < 2; ++kc)
    aQ[kc] = load_a_frag(qm + ((size_t)bh * NS + q0) * NDK + kc * 32, NDK, lane);

  v8f o[4];
#pragma unroll
  for (int dt = 0; dt < 4; ++dt) o[dt] = vzero8();
  float mrun[8], lrun[8];
#pragma unroll
  for (int r = 0; r < 8; ++r) { mrun[r] = -3.0e38f; lrun[r] = 0.f; }

  for (int kb = 0; kb < NS; kb += 64) {
    // S = (Q @ K^T) / 8 : K rows row-major == column-major of K^T
    v8f sf[4];
#pragma unroll
    for (int nt = 0; nt < 4; ++nt) {
      v8f s = vzero8();
#pragma unroll
      for (int kc = 0; kc < 2; ++kc)
        s = wmma_bf16(aQ[kc],
                      load_b_frag(km + ((size_t)bh * NS + kb + nt * 16) * NDK + kc * 32,
                                  NDK, lane),
                      s);
#pragma unroll
      for (int vg = 0; vg < 8; ++vg) s[vg] *= 0.125f;  // 1/sqrt(DK)
      sf[nt] = s;
    }
    // online softmax; row r of the C tile lives across one 16-lane half
#pragma unroll
    for (int r = 0; r < 8; ++r) {
      float tm = sf[0][r];
#pragma unroll
      for (int nt = 1; nt < 4; ++nt) tm = fmaxf(tm, sf[nt][r]);
      tm = hmax16(tm);
      float mnew = fmaxf(mrun[r], tm);
      float corr = __expf(mrun[r] - mnew);
      float rs = 0.f;
#pragma unroll
      for (int nt = 0; nt < 4; ++nt) {
        float p = __expf(sf[nt][r] - mnew);
        sf[nt][r] = p;
        rs += p;
      }
      rs = hsum16(rs);
      lrun[r] = lrun[r] * corr + rs;
      mrun[r] = mnew;
#pragma unroll
      for (int dt = 0; dt < 4; ++dt) o[dt][r] *= corr;
    }
    // P: C-layout -> LDS (row-major) -> A-fragments
#pragma unroll
    for (int nt = 0; nt < 4; ++nt)
#pragma unroll
      for (int vg = 0; vg < 8; ++vg)
        p_lds[wave][vg + rof][nt * 16 + lh] = f2bf(sf[nt][vg]);
    __syncthreads();
    v16bf aP[2];
#pragma unroll
    for (int kc = 0; kc < 2; ++kc)
      aP[kc] = load_a_frag(&p_lds[wave][0][kc * 32], 64, lane);
    // O += P @ V  (vT rows are V columns -> contiguous B-fragments)
#pragma unroll
    for (int dt = 0; dt < 4; ++dt)
#pragma unroll
      for (int kc = 0; kc < 2; ++kc)
        o[dt] = wmma_bf16(aP[kc],
                          load_b_frag(vT + ((size_t)bh * NDK + dt * 16) * NS + kb + kc * 32,
                                      NS, lane),
                          o[dt]);
    __syncthreads();
  }

  // epilogue: normalize, write [B, S, H*DK] fp32
  const int b = bh >> 4, h = bh & 15;
#pragma unroll
  for (int dt = 0; dt < 4; ++dt) {
    int gn = h * NDK + dt * 16 + lh;
#pragma unroll
    for (int vg = 0; vg < 8; ++vg) {
      int row = q0 + vg + rof;
      out[((size_t)b * NS + row) * NE + gn] = o[dt][vg] / lrun[vg];
    }
  }
}

// ---------------------------------------------------------------------------
extern "C" void kernel_launch(void* const* d_in, const int* in_sizes, int n_in,
                              void* d_out, int out_size, void* d_ws, size_t ws_size,
                              hipStream_t stream) {
  (void)in_sizes; (void)n_in; (void)out_size; (void)ws_size;
  const float* inp = (const float*)d_in[0];
  const float* Wq  = (const float*)d_in[1];
  const float* bq  = (const float*)d_in[2];
  const float* Wk  = (const float*)d_in[3];
  const float* bk  = (const float*)d_in[4];
  const float* Wv  = (const float*)d_in[5];
  const float* bv  = (const float*)d_in[6];
  const float* Wfq = (const float*)d_in[7];
  const float* bfq = (const float*)d_in[8];
  const float* Wfk = (const float*)d_in[9];
  const float* bfk = (const float*)d_in[10];
  const float* Wfg = (const float*)d_in[11];
  const float* bfg = (const float*)d_in[12];
  float* out = (float*)d_out;

  const size_t elems = (size_t)NB * NS * NE;  // 8,388,608 per tensor
  __bf16* qbuf = (__bf16*)d_ws;               // [B,H,S,DK] bf16 (modulated in place)
  __bf16* kbuf = qbuf + elems;                // [B,H,S,DK] bf16 (modulated in place)
  __bf16* vT   = kbuf + elems;                // [B,H,DK,S] bf16 (transposed V)

  qkv_proj_kernel<<<dim3(64, 8, 3), dim3(256), 0, stream>>>(
      inp, Wq, bq, Wk, bk, Wv, bv, qbuf, kbuf, vT);
  gate_kernel<<<dim3(1024), dim3(256), 0, stream>>>(
      Wfq, bfq, Wfk, bfk, Wfg, bfg, qbuf, kbuf);
  attn_kernel<<<dim3(1024), dim3(256), 0, stream>>>(qbuf, kbuf, vT, out);
}
